// GPRMamba2_55473797595987
// MI455X (gfx1250) — matrix operand order repaired and checked by reading, not compiled, not run on hardware
//
#include <hip/hip_runtime.h>
#include <hip/hip_bf16.h>
#include <math.h>

// ---- model constants ----
#define HGc 17
#define WGc 90
#define PHc 20
#define PWc 8
#define CINc 3
#define DMc 128
#define NHc 4
#define HDc 64
#define NSc 128
#define EXPc 256
#define LPc (HGc*WGc)          // 1530
#define TTc (4*LPc)            // 6120
#define BBc 8
#define PNc 772                // in-proj output width
#define PKc (CINc*PHc*PWc)     // 480
#define QCc 120                // scan chunk length
#define NCHc 51                // chunks per sequence (51*120 = 6120)

typedef __attribute__((ext_vector_type(16))) _Float16 v16h;
typedef __attribute__((ext_vector_type(8)))  _Float16 v8h;
typedef __attribute__((ext_vector_type(4)))  _Float16 v4h;
typedef __attribute__((ext_vector_type(8)))  float    v8f;
typedef __attribute__((ext_vector_type(4)))  unsigned int v4u;
typedef __attribute__((ext_vector_type(8)))  int      v8i;
typedef __attribute__((ext_vector_type(4)))  int      v4i;

union AFrag { v16h v; v8h h[2]; };

#define USE_TDM (__has_builtin(__builtin_amdgcn_tensor_load_to_lds) && (__clang_major__ == 22))

// ---------------------------------------------------------------------------
// WMMA GEMM: C[M,N] = A[M,K] @ W[Npad,K]^T (+bias) (+=C if accum)
// W zero-padded to Npad (mult of 64): divergence-free k-loop.
// Block = 8 waves x (16x64 strip each), all sharing one 64-row W strip.
// The W strip (64*K halfs, contiguous) is staged into LDS once per block by
// the Tensor Data Mover as a 1-D tile (D# per ISA 8.3/8.4), then B-fragments
// are served by ds_load; A-fragments stream from global.
// Fragment layouts as loaded here (contiguous k, A and B alike):
//   A (16x32 f16): lane m=lane&15; halves 0..7 -> K=k0+16*(lane>=16)+e,
//                  halves 8..15   -> K=k0+16*(lane>=16)+e
//   B (32x16 f16): lane n=lane&15; halves 0..15 -> K=k0+16*(lane>=16)+e
//   C/D (16x16 f32): elem r -> row m0 + r + 8*(lane>=16), col n0+(lane&15)
// ---------------------------------------------------------------------------
__global__ void k_wmma_gemm(const _Float16* __restrict__ A,
                            const _Float16* __restrict__ W,
                            const float* __restrict__ bias,
                            float* __restrict__ C,
                            int M, int N, int K, int ldc, int accum)
{
  extern __shared__ _Float16 Wlds[];          // 64*K halfs
  const int tid = threadIdx.x;
  const int n0 = blockIdx.x * 64;
  const int elems = 64 * K;                   // <= 30720, fits tile_dim0[15:0]

#if USE_TDM
  if ((tid >> 5) == 0) {                      // one wave issues the DMA
    unsigned long long ga = (unsigned long long)(uintptr_t)(W + (size_t)n0 * K);
    unsigned lds_lo = (unsigned)(uintptr_t)(void*)Wlds;
    v4u g0;
    g0.x = 1u;                                                  // count=1, load
    g0.y = lds_lo;                                              // lds_addr
    g0.z = (unsigned)(ga & 0xffffffffu);                        // gaddr[31:0]
    g0.w = (unsigned)((ga >> 32) & 0x01ffffffu) | (2u << 30);   // gaddr[56:32], type=2
    v8i g1;
    g1[0] = (1 << 16);                        // data_size=1 (2 bytes), no flags
    g1[1] = (elems & 0xffff) << 16;           // tensor_dim0[15:0]
    g1[2] = ((elems >> 16) & 0xffff) | (1 << 16); // tensor_dim0[31:16], tensor_dim1=1
    g1[3] = (elems & 0xffff) << 16;           // tile_dim0 (1-D tile)
    g1[4] = 0;                                // tile_dim1=0, tile_dim2=0 (unused)
    g1[5] = elems;                            // tensor_dim0_stride[31:0]
    g1[6] = 0;
    g1[7] = 0;
    v4i gz = {0, 0, 0, 0};
    __builtin_amdgcn_tensor_load_to_lds(g0, g1, gz, gz, 0);
    __builtin_amdgcn_s_wait_tensorcnt(0);
  }
#else
  {
    const v8h* src = (const v8h*)(W + (size_t)n0 * K);
    v8h* dst = (v8h*)Wlds;
    const int nv = elems / 8;
    for (int i = tid; i < nv; i += 256) dst[i] = src[i];
  }
#endif
  __syncthreads();

  const int lane = tid & 31;
  const int wave = tid >> 5;
  const int m0 = (blockIdx.y * 8 + wave) * 16;
  if (m0 >= M) return;                        // wave-uniform: EXEC stays full
  const int ln = lane & 15;
  const int hi = lane >> 4;                   // 0 or 1
  const _Float16* arow = A + (size_t)(m0 + ln) * K;
  const _Float16* w0   = Wlds + (size_t)ln * K;
  v8f c0 = {}, c1 = {}, c2 = {}, c3 = {};
  for (int k0 = 0; k0 < K; k0 += 32) {
    AFrag a;
    a.h[0] = *(const v8h*)(arow + k0 + hi * 16);
    a.h[1] = *(const v8h*)(arow + k0 + hi * 16 + 8);
    const int ko = k0 + hi * 16;
    v16h b0 = *(const v16h*)(w0 + ko);
    v16h b1 = *(const v16h*)(w0 + (size_t)16 * K + ko);
    v16h b2 = *(const v16h*)(w0 + (size_t)32 * K + ko);
    v16h b3 = *(const v16h*)(w0 + (size_t)48 * K + ko);
    c0 = __builtin_amdgcn_wmma_f32_16x16x32_f16(false, a.v, false, b0, (short)0, c0, false, false);
    c1 = __builtin_amdgcn_wmma_f32_16x16x32_f16(false, a.v, false, b1, (short)0, c1, false, false);
    c2 = __builtin_amdgcn_wmma_f32_16x16x32_f16(false, a.v, false, b2, (short)0, c2, false, false);
    c3 = __builtin_amdgcn_wmma_f32_16x16x32_f16(false, a.v, false, b3, (short)0, c3, false, false);
  }
  const int mr = m0 + hi * 8;
  v8f cc[4] = {c0, c1, c2, c3};
  #pragma unroll
  for (int j = 0; j < 4; ++j) {
    const int n = n0 + j * 16 + ln;
    if (n < N) {
      const float bv = bias ? bias[n] : 0.0f;
      #pragma unroll
      for (int r = 0; r < 8; ++r) {
        size_t idx = (size_t)(mr + r) * ldc + n;
        float val = cc[j][r] + bv;
        if (accum) val += C[idx];
        C[idx] = val;
      }
    }
  }
}

// ---- f32 -> f16 weight convert with zero row padding to Npad ----
__global__ void k_cvt_pad(const float* __restrict__ s, _Float16* __restrict__ d,
                          int N, int K, int Npad)
{
  int i = blockIdx.x * blockDim.x + threadIdx.x;
  if (i >= Npad * K) return;
  int n = i / K;
  d[i] = (n < N) ? (_Float16)s[i] : (_Float16)0.0f;
}

// ---- im2col for patch embed: x(B,CIN,340,720) -> A[B*1530, 480] f16 ----
__global__ void k_im2col(const float* __restrict__ x, _Float16* __restrict__ a)
{
  int idx = blockIdx.x * blockDim.x + threadIdx.x;
  if (idx >= BBc * LPc * PKc) return;
  int k = idx % PKc;       int row = idx / PKc;
  int b = row / LPc;       int t = row % LPc;
  int hg = t / WGc;        int wg = t % WGc;
  int c = k / (PHc*PWc);   int r = k % (PHc*PWc);
  int i = r / PWc;         int j = r % PWc;
  size_t src = (((size_t)(b*CINc + c) * (HGc*PHc)) + hg*PHc + i) * (size_t)(WGc*PWc)
               + wg*PWc + j;
  a[idx] = (_Float16)x[src];
}

// ---- build 4-stream sequence h(B, 6120, 128) from emb(B,1530,128) ----
__global__ void k_build_h(const float* __restrict__ emb, float* __restrict__ h)
{
  int idx = blockIdx.x * blockDim.x + threadIdx.x;
  if (idx >= BBc * TTc * DMc) return;
  int d = idx & (DMc-1);   int bt = idx >> 7;
  int t = bt % TTc;        int b = bt / TTc;
  int tl = t % LPc;        int seg = t / LPc;
  int src;
  if (seg == 0)      src = tl;
  else if (seg == 1) { int wg = tl / HGc, hg = tl % HGc; src = hg*WGc + wg; }
  else if (seg == 2) src = LPc - 1 - tl;
  else { int r = LPc - 1 - tl; int wg = r / HGc, hg = r % HGc; src = hg*WGc + wg; }
  h[idx] = emb[((size_t)(b*LPc + src)) * DMc + d];
}

// ---- rmsnorm over 128, one wave/row; out f16 (for WMMA A) or f32 ----
__global__ void k_rmsnorm128(const float* __restrict__ in, const float* __restrict__ w,
                             void* __restrict__ out, int rows, int out_half)
{
  int lane = threadIdx.x & 31, wave = threadIdx.x >> 5;
  int row = blockIdx.x * 8 + wave;
  if (row >= rows) return;
  const float4 xv = *(const float4*)(in + (size_t)row * DMc + lane * 4);
  float s = xv.x*xv.x + xv.y*xv.y + xv.z*xv.z + xv.w*xv.w;
  #pragma unroll
  for (int off = 16; off; off >>= 1) s += __shfl_xor(s, off, 32);
  float sc = rsqrtf(s * (1.0f/DMc) + 1e-5f);
  const float4 wv = *(const float4*)(w + lane * 4);
  float o0 = xv.x*sc*wv.x, o1 = xv.y*sc*wv.y, o2 = xv.z*sc*wv.z, o3 = xv.w*sc*wv.w;
  if (out_half) {
    v4h o; o.x = (_Float16)o0; o.y = (_Float16)o1; o.z = (_Float16)o2; o.w = (_Float16)o3;
    *(((v4h*)out) + (size_t)row * 32 + lane) = o;
  } else {
    float4 o; o.x = o0; o.y = o1; o.z = o2; o.w = o3;
    *(((float4*)out) + (size_t)row * 32 + lane) = o;
  }
}

// ---- causal depthwise conv1d (K=4) + silu over proj cols [256,768) ----
__global__ void k_conv_silu(const float* __restrict__ proj,
                            const float* __restrict__ cw, const float* __restrict__ cb,
                            float* __restrict__ xbcc)
{
  int idx = blockIdx.x * blockDim.x + threadIdx.x;
  if (idx >= BBc * TTc * 512) return;
  int c = idx & 511;  int bt = idx >> 9;
  int t = bt % TTc;   int b = bt / TTc;
  float acc = cb[c];
  #pragma unroll
  for (int k = 0; k < 4; ++k) {
    int ts = t - 3 + k;
    if (ts >= 0)
      acc += proj[((size_t)(b*TTc + ts)) * PNc + 256 + c] * cw[c*4 + k];
  }
  xbcc[(size_t)bt * 512 + c] = acc / (1.0f + expf(-acc));
}

// ---- dt precompute: dtv = softplus(dt_raw + bias), dA = exp(dtv * A) ----
__global__ void k_dt(const float* __restrict__ proj, const float* __restrict__ dtb,
                     const float* __restrict__ alog, float* __restrict__ dtv,
                     float* __restrict__ dAv)
{
  int idx = blockIdx.x * blockDim.x + threadIdx.x;
  if (idx >= BBc * TTc * NHc) return;
  int hh = idx & 3; int row = idx >> 2;
  float Ah = -expf(alog[hh]);
  float raw = proj[(size_t)row * PNc + 768 + hh] + dtb[hh];
  float v = (raw > 20.0f) ? raw : log1pf(expf(raw));
  dtv[idx] = v;
  dAv[idx] = expf(v * Ah);
}

// ===========================================================================
// Chunked SSD scan (3 passes). Decay is scalar per (t,head), so:
//   pass 1: per chunk, local state U_c (zero init) and decay product P_c
//   pass 2: 51-step prefix over chunks: S_in(c); ust is rewritten in place
//   pass 3: per chunk, replay recurrence from S_in(c), emit y
// State (64x128 f32) lives in VGPRs: 512 threads x 16 regs.
// ===========================================================================
__global__ void __launch_bounds__(512)
k_chunk_state(const float* __restrict__ xbcc, const float* __restrict__ dtv,
              const float* __restrict__ dAv, float* __restrict__ ust,
              float* __restrict__ pprod)
{
  const int bh = blockIdx.x / NCHc;   // (b*4 + h), 0..31
  const int c  = blockIdx.x % NCHc;
  const int b  = bh >> 2, hh = bh & 3;
  const int tid = threadIdx.x;
  const int p = tid >> 3, n0 = (tid & 7) * 16;
  __shared__ float Bs[NSc];
  float st[16];
  #pragma unroll
  for (int i = 0; i < 16; ++i) st[i] = 0.0f;
  float pr = 1.0f;
  const int t0 = c * QCc;
  for (int t = t0; t < t0 + QCc; ++t) {
    const size_t row = (size_t)b * TTc + t;
    const float dA = dAv[row*4 + hh];
    const float dt = dtv[row*4 + hh];
    const float* xr = xbcc + row * 512;
    float xv = xr[hh * HDc + p];
    if (tid < 128) Bs[tid] = xr[256 + tid];
    __syncthreads();
    const float dtx = dt * xv;
    #pragma unroll
    for (int i = 0; i < 16; ++i) st[i] = st[i] * dA + dtx * Bs[n0 + i];
    pr *= dA;
    __syncthreads();
  }
  float* up = ust + ((size_t)bh * NCHc + c) * 8192 + tid * 16;
  #pragma unroll
  for (int i = 0; i < 16; ++i) up[i] = st[i];
  if (tid == 0) pprod[bh * NCHc + c] = pr;
}

__global__ void __launch_bounds__(512)
k_state_prefix(float* __restrict__ ust, const float* __restrict__ pprod)
{
  const int bh = blockIdx.x;
  const int tid = threadIdx.x;
  float st[16];
  #pragma unroll
  for (int i = 0; i < 16; ++i) st[i] = 0.0f;
  for (int c = 0; c < NCHc; ++c) {
    float* up = ust + ((size_t)bh * NCHc + c) * 8192 + tid * 16;
    const float Pc = pprod[bh * NCHc + c];
    #pragma unroll
    for (int i = 0; i < 16; ++i) {
      float u = up[i];
      up[i] = st[i];                 // S_in(c) for pass 3
      st[i] = Pc * st[i] + u;
    }
  }
}

__global__ void __launch_bounds__(512)
k_chunk_scan(const float* __restrict__ xbcc, const float* __restrict__ dtv,
             const float* __restrict__ dAv, const float* __restrict__ ust,
             const float* __restrict__ dvec, float* __restrict__ y)
{
  const int bh = blockIdx.x / NCHc;
  const int c  = blockIdx.x % NCHc;
  const int b  = bh >> 2, hh = bh & 3;
  const int tid = threadIdx.x;
  const int p = tid >> 3, ng = tid & 7, n0 = ng * 16;
  const float Dph = dvec[hh];
  __shared__ float Bs[NSc], Cs[NSc];
  const float* up = ust + ((size_t)bh * NCHc + c) * 8192 + tid * 16;
  float st[16];
  #pragma unroll
  for (int i = 0; i < 16; ++i) st[i] = up[i];
  const int t0 = c * QCc;
  for (int t = t0; t < t0 + QCc; ++t) {
    const size_t row = (size_t)b * TTc + t;
    const float dA = dAv[row*4 + hh];
    const float dt = dtv[row*4 + hh];
    const float* xr = xbcc + row * 512;
    float xv = xr[hh * HDc + p];
    if (tid < 128)      Bs[tid]       = xr[256 + tid];
    else if (tid < 256) Cs[tid - 128] = xr[384 + (tid - 128)];
    if (t + 1 < t0 + QCc)
      __builtin_prefetch(xr + 512 + tid, 0, 1);   // next row -> global_prefetch_b8
    __syncthreads();
    const float dtx = dt * xv;
    float acc = 0.0f;
    #pragma unroll
    for (int i = 0; i < 16; ++i) {
      float s = st[i] * dA + dtx * Bs[n0 + i];
      st[i] = s;
      acc += s * Cs[n0 + i];
    }
    acc += __shfl_xor(acc, 4, 8);
    acc += __shfl_xor(acc, 2, 8);
    acc += __shfl_xor(acc, 1, 8);
    if (ng == 0) y[row * EXPc + hh * HDc + p] = acc + Dph * xv;
    __syncthreads();
  }
}

// ---- y * silu(gate) then rmsnorm over 256 with gn_w -> f16 ----
__global__ void k_gate_rmsnorm(const float* __restrict__ y, const float* __restrict__ proj,
                               const float* __restrict__ gw, _Float16* __restrict__ out,
                               int rows)
{
  int lane = threadIdx.x & 31, wave = threadIdx.x >> 5;
  int row = blockIdx.x * 8 + wave;
  if (row >= rows) return;
  const float* yr = y + (size_t)row * EXPc;
  const float* gr = proj + (size_t)row * PNc;   // gate = first 256 cols of proj
  int c0 = lane * 8;
  float v[8]; float s = 0.0f;
  #pragma unroll
  for (int i = 0; i < 8; ++i) {
    float g = gr[c0 + i];
    float val = yr[c0 + i] * (g / (1.0f + expf(-g)));
    v[i] = val; s += val * val;
  }
  #pragma unroll
  for (int off = 16; off; off >>= 1) s += __shfl_xor(s, off, 32);
  float sc = rsqrtf(s * (1.0f/EXPc) + 1e-5f);
  #pragma unroll
  for (int i = 0; i < 8; ++i)
    out[(size_t)row * EXPc + c0 + i] = (_Float16)(v[i] * sc * gw[c0 + i]);
}

// ---- recombine the 4 streams into xm(B,1530,128) f16 ----
__global__ void k_combine(const float* __restrict__ hf, _Float16* __restrict__ xm)
{
  int idx = blockIdx.x * blockDim.x + threadIdx.x;
  if (idx >= BBc * LPc * DMc) return;
  int d = idx & (DMc-1);  int bt = idx >> 7;
  int b = bt / LPc;       int t = bt % LPc;
  int hg = t / WGc, wg = t % WGc;
  size_t base = (size_t)b * TTc;
  float v = hf[(base + t) * DMc + d];
  v += hf[(base + LPc   + wg*HGc + hg) * DMc + d];
  v += hf[(base + 2*LPc + (LPc - 1 - t)) * DMc + d];
  v += hf[(base + 3*LPc + (LPc - 1 - (wg*HGc + hg))) * DMc + d];
  xm[idx] = (_Float16)v;
}

// ---- exact gelu, f32 -> f16 ----
__global__ void k_gelu_f16(const float* __restrict__ s, _Float16* __restrict__ d, int n)
{
  int i = blockIdx.x * blockDim.x + threadIdx.x;
  if (i >= n) return;
  float v = s[i];
  d[i] = (_Float16)(0.5f * v * (1.0f + erff(v * 0.70710678118654752f)));
}

// ---- scatter z2(B*1530, 480) -> out(B, 3, 340, 720) ----
__global__ void k_scatter(const float* __restrict__ z2, float* __restrict__ out, int n)
{
  int idx = blockIdx.x * blockDim.x + threadIdx.x;
  if (idx >= n) return;
  int j2 = idx % (WGc*PWc);  int rest = idx / (WGc*PWc);
  int i2 = rest % (HGc*PHc); rest /= (HGc*PHc);
  int c = rest % CINc;       int b = rest / CINc;
  int hg = i2 / PHc, i = i2 % PHc, wg = j2 / PWc, j = j2 % PWc;
  out[idx] = z2[((size_t)(b*LPc + hg*WGc + wg)) * PKc + c*(PHc*PWc) + i*PWc + j];
}

// ---------------------------------------------------------------------------
extern "C" void kernel_launch(void* const* d_in, const int* in_sizes, int n_in,
                              void* d_out, int out_size, void* d_ws, size_t ws_size,
                              hipStream_t stream)
{
  (void)in_sizes; (void)n_in; (void)ws_size;
  const float* x       = (const float*)d_in[0];
  const float* W_pe    = (const float*)d_in[1];
  const float* b_pe    = (const float*)d_in[2];
  const float* norm_w  = (const float*)d_in[3];
  const float* in_w    = (const float*)d_in[4];
  const float* in_b    = (const float*)d_in[5];
  const float* conv_w  = (const float*)d_in[6];
  const float* conv_b  = (const float*)d_in[7];
  const float* dt_bias = (const float*)d_in[8];
  const float* A_log   = (const float*)d_in[9];
  const float* Dv      = (const float*)d_in[10];
  const float* gn_w    = (const float*)d_in[11];
  const float* out_w   = (const float*)d_in[12];
  const float* out_b   = (const float*)d_in[13];
  const float* normf_w = (const float*)d_in[14];
  const float* r1_w    = (const float*)d_in[15];
  const float* r1_b    = (const float*)d_in[16];
  const float* r2_w    = (const float*)d_in[17];
  const float* r2_b    = (const float*)d_in[18];
  float* outp = (float*)d_out;

  char* ws = (char*)d_ws;
  size_t off = 0;
  auto alloc = [&](size_t bytes) -> char* {
    off = (off + 255) & ~(size_t)255;
    char* p = ws + off; off += bytes; return p;
  };
  auto cdiv = [](int a, int b) { return (a + b - 1) / b; };

  const int BT = BBc * TTc;   // 48960 rows (full 4-stream sequence)
  const int BL = BBc * LPc;   // 12240 rows (single stream)
  const int INWP = 832;       // in_w N padded to mult of 64
  const int R2P  = 512;       // r2_w N padded

  _Float16* Ape    = (_Float16*)alloc((size_t)BL * PKc * 2);
  _Float16* Wpe16  = (_Float16*)alloc((size_t)DMc * PKc * 2);
  _Float16* inw16  = (_Float16*)alloc((size_t)4 * INWP * DMc * 2);
  _Float16* outw16 = (_Float16*)alloc((size_t)4 * DMc * EXPc * 2);
  _Float16* r1w16  = (_Float16*)alloc((size_t)DMc * DMc * 2);
  _Float16* r2w16  = (_Float16*)alloc((size_t)R2P * DMc * 2);
  float*    emb    = (float*)   alloc((size_t)BL * DMc * 4);
  float*    h      = (float*)   alloc((size_t)BT * DMc * 4);
  _Float16* hn16   = (_Float16*)alloc((size_t)BT * DMc * 2);
  float*    proj   = (float*)   alloc((size_t)BT * PNc * 4);
  float*    xbcc   = (float*)   alloc((size_t)BT * 512 * 4);
  float*    yssd   = (float*)   alloc((size_t)BT * EXPc * 4);
  _Float16* yg16   = (_Float16*)alloc((size_t)BT * EXPc * 2);
  float*    dtvb   = (float*)   alloc((size_t)BT * NHc * 4);
  float*    dAvb   = (float*)   alloc((size_t)BT * NHc * 4);
  float*    ust    = (float*)   alloc((size_t)32 * NCHc * 8192 * 4);
  float*    pprod  = (float*)   alloc((size_t)32 * NCHc * 4);
  // post-layer stages reuse earlier (now-dead) buffers
  float*    hf   = yssd;
  _Float16* xm16 = yg16;
  float*    z1   = xbcc;
  _Float16* z1g  = hn16;
  float*    z2   = proj;

  const int thr = 256;
  auto gemm = [&](const _Float16* A, const _Float16* W, const float* bias, float* C,
                  int M, int N, int K, int Npad, int ldc, int accum) {
    dim3 g(Npad / 64, cdiv(M / 16, 8));
    size_t shmem = (size_t)64 * K * 2;     // W strip staged in LDS
    k_wmma_gemm<<<g, thr, shmem, stream>>>(A, W, bias, C, M, N, K, ldc, accum);
  };

  // weight conversions with zero padding (deterministic, every call)
  k_cvt_pad<<<cdiv(DMc*PKc, thr), thr, 0, stream>>>(W_pe, Wpe16, DMc, PKc, DMc);
  for (int l = 0; l < 4; ++l)
    k_cvt_pad<<<cdiv(INWP*DMc, thr), thr, 0, stream>>>(in_w + (size_t)l*PNc*DMc,
                                                       inw16 + (size_t)l*INWP*DMc,
                                                       PNc, DMc, INWP);
  k_cvt_pad<<<cdiv(4*DMc*EXPc, thr), thr, 0, stream>>>(out_w, outw16, 4*DMc, EXPc, 4*DMc);
  k_cvt_pad<<<cdiv(DMc*DMc, thr), thr, 0, stream>>>(r1_w, r1w16, DMc, DMc, DMc);
  k_cvt_pad<<<cdiv(R2P*DMc, thr), thr, 0, stream>>>(r2_w, r2w16, PKc, DMc, R2P);

  // patch embedding + 4-stream sequence
  k_im2col<<<cdiv(BL*PKc, thr), thr, 0, stream>>>(x, Ape);
  gemm(Ape, Wpe16, b_pe, emb, BL, DMc, PKc, 128, DMc, 0);
  k_build_h<<<cdiv(BT*DMc, thr), thr, 0, stream>>>(emb, h);

  // mamba layers
  for (int l = 0; l < 4; ++l) {
    k_rmsnorm128<<<cdiv(BT, 8), thr, 0, stream>>>(h, norm_w + l*DMc, hn16, BT, 1);
    gemm(hn16, inw16 + (size_t)l*INWP*DMc, in_b + l*PNc, proj, BT, PNc, DMc, INWP, PNc, 0);
    k_conv_silu<<<cdiv(BT*512, thr), thr, 0, stream>>>(proj, conv_w + l*512*4,
                                                       conv_b + l*512, xbcc);
    k_dt<<<cdiv(BT*NHc, thr), thr, 0, stream>>>(proj, dt_bias + l*NHc, A_log + l*NHc,
                                                dtvb, dAvb);
    k_chunk_state<<<32*NCHc, 512, 0, stream>>>(xbcc, dtvb, dAvb, ust, pprod);
    k_state_prefix<<<32, 512, 0, stream>>>(ust, pprod);
    k_chunk_scan<<<32*NCHc, 512, 0, stream>>>(xbcc, dtvb, dAvb, ust, Dv + l*NHc, yssd);
    k_gate_rmsnorm<<<cdiv(BT, 8), thr, 0, stream>>>(yssd, proj, gn_w + l*EXPc, yg16, BT);
    gemm(yg16, outw16 + (size_t)l*DMc*EXPc, out_b + l*DMc, h, BT, DMc, EXPc, 128, DMc, 1);
  }

  // final norm, stream recombine, MLP head
  k_rmsnorm128<<<cdiv(BT, 8), thr, 0, stream>>>(h, normf_w, hf, BT, 0);
  k_combine<<<cdiv(BL*DMc, thr), thr, 0, stream>>>(hf, xm16);
  gemm(xm16, r1w16, r1_b, z1, BL, DMc, DMc, 128, DMc, 0);
  k_gelu_f16<<<cdiv(BL*DMc, thr), thr, 0, stream>>>(z1, z1g, BL*DMc);
  gemm(z1g, r2w16, r2_b, z2, BL, PKc, DMc, R2P, PKc, 0);
  k_scatter<<<cdiv(out_size, thr), thr, 0, stream>>>(z2, outp, out_size);
}
